// GDKLayer_38457137168877
// MI455X (gfx1250) — matrix operand order repaired and checked by reading, not compiled, not run on hardware
//
#include <hip/hip_runtime.h>
#include <stdint.h>

#define B_  8
#define C_  512
#define T_  4096
#define GH_ 128
#define MH_ 2048
#define EPS_ 1e-6f

// WMMA tiling
#define NT1 (MH_/16)   // 128 n-tiles, GEMM1
#define KT1 (C_/32)    // 16 k-steps,  GEMM1
#define NT2 (C_/16)    // 32 n-tiles,  GEMM2
#define KT2 (MH_/32)   // 64 k-steps,  GEMM2
#define MROWS 32       // rows (t positions) per workgroup = 2 M-tiles

typedef __attribute__((ext_vector_type(16))) __bf16 v16bf;
typedef __attribute__((ext_vector_type(8)))  float  v8f;

struct alignas(16) Q4 { unsigned int x[4]; };
union V16U { Q4 q[2]; v16bf v; };

__device__ __forceinline__ unsigned short f2bf(float f) {
    union { float f; unsigned int u; } c; c.f = f;
    unsigned int u = c.u;
    unsigned int r = (u + 0x7FFFu + ((u >> 16) & 1u)) >> 16;
    return (unsigned short)r;
}

__device__ __forceinline__ v16bf load_bf16x16(const unsigned short* p0,
                                              const unsigned short* p1) {
    V16U u;
    u.q[0] = *(const Q4*)p0;
    u.q[1] = *(const Q4*)p1;
    return u.v;
}

__device__ __forceinline__ float gelu_exact(float x) {
    return 0.5f * x * (1.0f + erff(x * 0.70710678118654752f));
}

// Async copy of one lane's 32B (two b128) from global into LDS (ASYNCcnt +2).
__device__ __forceinline__ void async_copy_lane32(unsigned lds_byte,
                                                  const unsigned short* gptr) {
    unsigned long long ga = (unsigned long long)gptr;
    asm volatile(
        "global_load_async_to_lds_b128 %0, %1, off\n\t"
        "global_load_async_to_lds_b128 %0, %1, off offset:16"
        :: "v"(lds_byte), "v"(ga) : "memory");
}

__device__ __forceinline__ unsigned lds_off(const void* p) {
    return (unsigned)(unsigned long long)p;   // low 32 bits of generic = LDS offset
}

// ---------------- weight packing into WMMA B-operand layout -----------------
// pw[((kt*NT + nt)*32 + lane)*16 + e] = bf16( W[nt*16+(lane&15)][kt*32 + (lane>>4)*8 + ((e&8)<<1) + (e&7)] )
__global__ void pack_w1_kernel(const float* __restrict__ w, unsigned short* __restrict__ pw) {
    int idx = blockIdx.x * 256 + threadIdx.x;      // < MH_*C_
    int e = idx & 15;
    int lane = (idx >> 4) & 31;
    int rest = idx >> 9;
    int nt = rest % NT1;
    int kt = rest / NT1;
    pw[idx] = f2bf(w[(nt * 16 + (lane & 15)) * C_ + (kt * 32 + (lane >> 4) * 8 + ((e & 8) << 1) + (e & 7))]);
}

__global__ void pack_w2_kernel(const float* __restrict__ w, unsigned short* __restrict__ pw) {
    int idx = blockIdx.x * 256 + threadIdx.x;      // < C_*MH_
    int e = idx & 15;
    int lane = (idx >> 4) & 31;
    int rest = idx >> 9;
    int nt = rest % NT2;
    int kt = rest / NT2;
    pw[idx] = f2bf(w[(nt * 16 + (lane & 15)) * MH_ + (kt * 32 + (lane >> 4) * 8 + ((e & 8) << 1) + (e & 7))]);
}

// ---------------- LayerNorm 1 over channel dim ------------------------------
__global__ void ln1_kernel(const float* __restrict__ x, const float* __restrict__ g,
                           const float* __restrict__ b, float* __restrict__ nx) {
    int t = blockIdx.x * 256 + threadIdx.x;
    int bb = blockIdx.y;
    size_t base = (size_t)bb * C_ * T_ + t;
    float s = 0.f, s2 = 0.f;
    for (int c = 0; c < C_; ++c) {
        float v = x[base + (size_t)c * T_];
        s += v; s2 += v * v;
    }
    float mu = s * (1.0f / C_);
    float var = s2 * (1.0f / C_) - mu * mu;
    float rs = rsqrtf(var + EPS_);
    for (int c = 0; c < C_; ++c) {
        float v = x[base + (size_t)c * T_];
        nx[base + (size_t)c * T_] = (v - mu) * rs * g[c] + b[c];
    }
}

// ---------------- masked global-average partial (sum over T) ----------------
__global__ void gctx_kernel(const float* __restrict__ nx, const uint8_t* __restrict__ mask,
                            float* __restrict__ gctx) {
    __shared__ float red[256];
    int bc = blockIdx.x;                  // b*C_ + c
    int bb = bc / C_;
    float s = 0.f;
    for (int t = threadIdx.x; t < T_; t += 256) {
        float mf = mask[bb * T_ + t] ? 1.0f : 0.0f;
        s += nx[(size_t)bc * T_ + t] * mf;
    }
    red[threadIdx.x] = s;
    __syncthreads();
    for (int off = 128; off > 0; off >>= 1) {
        if (threadIdx.x < off) red[threadIdx.x] += red[threadIdx.x + off];
        __syncthreads();
    }
    if (threadIdx.x == 0) gctx[bc] = red[0];
}

// ---------------- gate MLP: 512 -> 128 -> 2 -> softmax ----------------------
__global__ void gate_kernel(const float* __restrict__ gctx, const uint8_t* __restrict__ mask,
                            const float* __restrict__ g1w, const float* __restrict__ g1b,
                            const float* __restrict__ g2w, const float* __restrict__ g2b,
                            float* __restrict__ gw) {
    __shared__ float s_h[GH_];
    __shared__ float s_l[2];
    __shared__ float s_scale;
    int tid = threadIdx.x;
    for (int bb = 0; bb < B_; ++bb) {
        if (tid == 0) {
            float v = 0.f;
            for (int t = 0; t < T_; ++t) v += mask[bb * T_ + t] ? 1.0f : 0.0f;
            s_scale = 1.0f / fmaxf(v, 1.0f);
        }
        __syncthreads();
        if (tid < GH_) {
            float acc = 0.f;
            for (int c = 0; c < C_; ++c) acc += gctx[bb * C_ + c] * g1w[tid * C_ + c];
            s_h[tid] = gelu_exact(acc * s_scale + g1b[tid]);
        }
        __syncthreads();
        if (tid < 2) {
            float acc = g2b[tid];
            for (int h = 0; h < GH_; ++h) acc += s_h[h] * g2w[tid * GH_ + h];
            s_l[tid] = acc;
        }
        __syncthreads();
        if (tid == 0) {
            float mx = fmaxf(s_l[0], s_l[1]);
            float e0 = expf(s_l[0] - mx), e1 = expf(s_l[1] - mx);
            float inv = 1.0f / (e0 + e1);
            gw[bb * 2 + 0] = e0 * inv;
            gw[bb * 2 + 1] = e1 * inv;
        }
        __syncthreads();
    }
}

// ---------------- depthwise convs + gated fuse + main_out -------------------
__global__ void conv_main_kernel(const float* __restrict__ x, const uint8_t* __restrict__ mask,
                                 const float* __restrict__ nx,
                                 const float* __restrict__ psi_w, const float* __restrict__ psi_b,
                                 const float* __restrict__ wgt_w, const float* __restrict__ wgt_b,
                                 const float* __restrict__ ak3_w, const float* __restrict__ ak3_b,
                                 const float* __restrict__ ak7_w, const float* __restrict__ ak7_b,
                                 const float* __restrict__ gw, float* __restrict__ main_) {
    int t = blockIdx.x * 256 + threadIdx.x;
    int c = blockIdx.y;
    int bb = blockIdx.z;
    size_t base = (size_t)bb * C_ * T_ + (size_t)c * T_;
    float nv[7];
#pragma unroll
    for (int j = 0; j < 7; ++j) {
        int tt = t + j - 3;
        nv[j] = (tt >= 0 && tt < T_) ? nx[base + tt] : 0.0f;
    }
    float psi = psi_w[c * 3] * nv[2] + psi_w[c * 3 + 1] * nv[3] + psi_w[c * 3 + 2] * nv[4] + psi_b[c];
    float wgt = wgt_w[c * 3] * nv[2] + wgt_w[c * 3 + 1] * nv[3] + wgt_w[c * 3 + 2] * nv[4] + wgt_b[c];
    float a3  = ak3_w[c * 3] * nv[2] + ak3_w[c * 3 + 1] * nv[3] + ak3_w[c * 3 + 2] * nv[4] + ak3_b[c];
    float a7 = ak7_b[c];
#pragma unroll
    for (int j = 0; j < 7; ++j) a7 += ak7_w[c * 7 + j] * nv[j];
    float fused = gw[bb * 2] * a3 + gw[bb * 2 + 1] * a7;
    float mf = mask[bb * T_ + t] ? 1.0f : 0.0f;
    main_[base + t] = x[base + t] * mf + fmaxf(wgt + fused, 0.0f) * psi;
}

// ---------------- LayerNorm 2 statistics ------------------------------------
__global__ void ln2stats_kernel(const float* __restrict__ main_, float* __restrict__ mu2,
                                float* __restrict__ rstd2) {
    int t = blockIdx.x * 256 + threadIdx.x;
    int bb = blockIdx.y;
    size_t base = (size_t)bb * C_ * T_ + t;
    float s = 0.f, s2 = 0.f;
    for (int c = 0; c < C_; ++c) {
        float v = main_[base + (size_t)c * T_];
        s += v; s2 += v * v;
    }
    float mu = s * (1.0f / C_);
    float var = s2 * (1.0f / C_) - mu * mu;
    mu2[bb * T_ + t] = mu;
    rstd2[bb * T_ + t] = rsqrtf(var + EPS_);
}

// ---------------- fused channel MLP via bf16 WMMA + async-LDS B pipe --------
// One WG = 32 rows (2 M-tiles), 8 waves. Each wave handles 2 n-tiles per
// k-loop. B operands are staged one k-step ahead into a wave-private LDS
// double buffer with GLOBAL_LOAD_ASYNC_TO_LDS_B128 (ASYNCcnt), so the WMMA
// stream only ever waits on LDS latency, not L2.
// GEMM1: [32x512]x[512x2048] -> GELU -> LDS bf16; GEMM2: [32x2048]x[2048x512].
__global__ void __launch_bounds__(256)
mlp_fused_kernel(const float* __restrict__ main_, const float* __restrict__ mu2,
                 const float* __restrict__ rstd2, const float* __restrict__ ln2g,
                 const float* __restrict__ ln2b, const unsigned short* __restrict__ pw1,
                 const float* __restrict__ m1b, const unsigned short* __restrict__ pw2,
                 const float* __restrict__ m2b, float* __restrict__ out) {
    __shared__ alignas(16) unsigned short s_a[MROWS * C_];       // 32KB
    __shared__ alignas(16) unsigned short s_h[MROWS * MH_];      // 128KB
    __shared__ alignas(16) unsigned short s_b[8][2][2][512];     // 32KB: B double buffer

    const int tid = threadIdx.x;
    const int tile = blockIdx.x;               // 0 .. B_*T_/MROWS - 1
    const int t0 = (tile % (T_ / MROWS)) * MROWS;
    const int bb = tile / (T_ / MROWS);
    const size_t gbase = (size_t)bb * C_ * T_;

    // Stage A-tile: normalize + convert to bf16, row-major [m][c].
    for (int i = tid; i < MROWS * C_; i += 256) {
        int m = i & (MROWS - 1), c = i / MROWS;
        int t = t0 + m;
        float v = main_[gbase + (size_t)c * T_ + t];
        float nrm = (v - mu2[bb * T_ + t]) * rstd2[bb * T_ + t] * ln2g[c] + ln2b[c];
        s_a[m * C_ + c] = f2bf(nrm);
    }
    __syncthreads();

    const int lane = tid & 31;
    const int wave = tid >> 5;
    const int m_l = lane & 15;
    const int half = lane >> 4;
    const int n_l = lane & 15;

    const unsigned sb0[2] = { lds_off(&s_b[wave][0][0][lane * 16]),
                              lds_off(&s_b[wave][1][0][lane * 16]) };
    const unsigned sb1[2] = { lds_off(&s_b[wave][0][1][lane * 16]),
                              lds_off(&s_b[wave][1][1][lane * 16]) };

    // ---- GEMM1: each wave owns 16 n-tiles, processed as 8 pairs ----
    for (int ntp = 0; ntp < 8; ++ntp) {
        const int nt0 = wave * 16 + ntp * 2;
        const int nt1 = nt0 + 1;
        v8f acc00 = {}, acc10 = {};
        v8f acc01 = {}, acc11 = {};
        // prologue: stage kt=0 into buffer 0
        async_copy_lane32(sb0[0], pw1 + ((size_t)(0 * NT1 + nt0) * 32 + lane) * 16);
        async_copy_lane32(sb1[0], pw1 + ((size_t)(0 * NT1 + nt1) * 32 + lane) * 16);
        for (int kt = 0; kt < KT1; ++kt) {
            const int st = kt & 1;
            if (kt + 1 < KT1) {
                async_copy_lane32(sb0[st ^ 1],
                    pw1 + ((size_t)((kt + 1) * NT1 + nt0) * 32 + lane) * 16);
                async_copy_lane32(sb1[st ^ 1],
                    pw1 + ((size_t)((kt + 1) * NT1 + nt1) * 32 + lane) * 16);
                asm volatile("s_wait_asynccnt 0x4" ::: "memory");
            } else {
                asm volatile("s_wait_asynccnt 0x0" ::: "memory");
            }
            const unsigned short* c0 = &s_b[wave][st][0][lane * 16];
            const unsigned short* c1 = &s_b[wave][st][1][lane * 16];
            v16bf b0 = load_bf16x16(c0, c0 + 8);
            v16bf b1 = load_bf16x16(c1, c1 + 8);
            const int k0 = kt * 32 + half * 8;
            v16bf a0 = load_bf16x16(&s_a[m_l * C_ + k0], &s_a[m_l * C_ + k0 + 16]);
            v16bf a1 = load_bf16x16(&s_a[(16 + m_l) * C_ + k0], &s_a[(16 + m_l) * C_ + k0 + 16]);
            acc00 = __builtin_amdgcn_wmma_f32_16x16x32_bf16(false, a0, false, b0, (short)0, acc00, false, false);
            acc10 = __builtin_amdgcn_wmma_f32_16x16x32_bf16(false, a1, false, b0, (short)0, acc10, false, false);
            acc01 = __builtin_amdgcn_wmma_f32_16x16x32_bf16(false, a0, false, b1, (short)0, acc01, false, false);
            acc11 = __builtin_amdgcn_wmma_f32_16x16x32_bf16(false, a1, false, b1, (short)0, acc11, false, false);
        }
        const int hcol0 = nt0 * 16 + n_l;
        const int hcol1 = nt1 * 16 + n_l;
        const float bias0 = m1b[hcol0];
        const float bias1 = m1b[hcol1];
#pragma unroll
        for (int r = 0; r < 8; ++r) {
            int m = r + half * 8;
            s_h[m * MH_ + hcol0] = f2bf(gelu_exact(acc00[r] + bias0));
            s_h[(16 + m) * MH_ + hcol0] = f2bf(gelu_exact(acc10[r] + bias0));
            s_h[m * MH_ + hcol1] = f2bf(gelu_exact(acc01[r] + bias1));
            s_h[(16 + m) * MH_ + hcol1] = f2bf(gelu_exact(acc11[r] + bias1));
        }
    }
    __syncthreads();

    // ---- GEMM2: each wave owns 4 n-tiles, processed as 2 pairs ----
    for (int ntp = 0; ntp < 2; ++ntp) {
        const int nt0 = wave * 4 + ntp * 2;
        const int nt1 = nt0 + 1;
        v8f acc00 = {}, acc10 = {};
        v8f acc01 = {}, acc11 = {};
        async_copy_lane32(sb0[0], pw2 + ((size_t)(0 * NT2 + nt0) * 32 + lane) * 16);
        async_copy_lane32(sb1[0], pw2 + ((size_t)(0 * NT2 + nt1) * 32 + lane) * 16);
        for (int kt = 0; kt < KT2; ++kt) {
            const int st = kt & 1;
            if (kt + 1 < KT2) {
                async_copy_lane32(sb0[st ^ 1],
                    pw2 + ((size_t)((kt + 1) * NT2 + nt0) * 32 + lane) * 16);
                async_copy_lane32(sb1[st ^ 1],
                    pw2 + ((size_t)((kt + 1) * NT2 + nt1) * 32 + lane) * 16);
                asm volatile("s_wait_asynccnt 0x4" ::: "memory");
            } else {
                asm volatile("s_wait_asynccnt 0x0" ::: "memory");
            }
            const unsigned short* c0 = &s_b[wave][st][0][lane * 16];
            const unsigned short* c1 = &s_b[wave][st][1][lane * 16];
            v16bf b0 = load_bf16x16(c0, c0 + 8);
            v16bf b1 = load_bf16x16(c1, c1 + 8);
            const int k0 = kt * 32 + half * 8;
            v16bf a0 = load_bf16x16(&s_h[m_l * MH_ + k0], &s_h[m_l * MH_ + k0 + 16]);
            v16bf a1 = load_bf16x16(&s_h[(16 + m_l) * MH_ + k0], &s_h[(16 + m_l) * MH_ + k0 + 16]);
            acc00 = __builtin_amdgcn_wmma_f32_16x16x32_bf16(false, a0, false, b0, (short)0, acc00, false, false);
            acc10 = __builtin_amdgcn_wmma_f32_16x16x32_bf16(false, a1, false, b0, (short)0, acc10, false, false);
            acc01 = __builtin_amdgcn_wmma_f32_16x16x32_bf16(false, a0, false, b1, (short)0, acc01, false, false);
            acc11 = __builtin_amdgcn_wmma_f32_16x16x32_bf16(false, a1, false, b1, (short)0, acc11, false, false);
        }
        const int ccol0 = nt0 * 16 + n_l;
        const int ccol1 = nt1 * 16 + n_l;
        const float bias0 = m2b[ccol0];
        const float bias1 = m2b[ccol1];
#pragma unroll
        for (int r = 0; r < 8; ++r) {
            int m = r + half * 8;
            size_t offA0 = gbase + (size_t)ccol0 * T_ + (t0 + m);
            size_t offA1 = gbase + (size_t)ccol0 * T_ + (t0 + 16 + m);
            size_t offB0 = gbase + (size_t)ccol1 * T_ + (t0 + m);
            size_t offB1 = gbase + (size_t)ccol1 * T_ + (t0 + 16 + m);
            out[offA0] = acc00[r] + bias0 + main_[offA0];
            out[offA1] = acc10[r] + bias0 + main_[offA1];
            out[offB0] = acc01[r] + bias1 + main_[offB0];
            out[offB1] = acc11[r] + bias1 + main_[offB1];
        }
    }
}

// ---------------- mask passthrough to output tail ---------------------------
__global__ void mask_out_kernel(const uint8_t* __restrict__ mask, float* __restrict__ out_tail) {
    int i = blockIdx.x * 256 + threadIdx.x;
    if (i < B_ * T_) out_tail[i] = mask[i] ? 1.0f : 0.0f;
}

extern "C" void kernel_launch(void* const* d_in, const int* in_sizes, int n_in,
                              void* d_out, int out_size, void* d_ws, size_t ws_size,
                              hipStream_t stream) {
    const float*   x     = (const float*)d_in[0];
    const uint8_t* mask  = (const uint8_t*)d_in[1];
    const float*   ln1g  = (const float*)d_in[2];
    const float*   ln1b  = (const float*)d_in[3];
    const float*   ln2g  = (const float*)d_in[4];
    const float*   ln2b  = (const float*)d_in[5];
    const float*   psi_w = (const float*)d_in[6];
    const float*   psi_b = (const float*)d_in[7];
    const float*   wgt_w = (const float*)d_in[8];
    const float*   wgt_b = (const float*)d_in[9];
    const float*   ak3_w = (const float*)d_in[10];
    const float*   ak3_b = (const float*)d_in[11];
    const float*   ak7_w = (const float*)d_in[12];
    const float*   ak7_b = (const float*)d_in[13];
    const float*   g1w   = (const float*)d_in[14];
    const float*   g1b   = (const float*)d_in[15];
    const float*   g2w   = (const float*)d_in[16];
    const float*   g2b   = (const float*)d_in[17];
    const float*   m1w   = (const float*)d_in[18];
    const float*   m1b   = (const float*)d_in[19];
    const float*   m2w   = (const float*)d_in[20];
    const float*   m2b   = (const float*)d_in[21];

    float* out = (float*)d_out;

    // Workspace layout (~132.3 MB total)
    const size_t SZ = (size_t)B_ * C_ * T_;              // 16,777,216 floats
    float* nx    = (float*)d_ws;
    float* mainb = nx + SZ;
    float* gctx  = mainb + SZ;                           // B*C
    float* gw    = gctx + (size_t)B_ * C_;               // B*2
    float* mu2   = gw + (size_t)B_ * 2;                  // B*T
    float* rstd2 = mu2 + (size_t)B_ * T_;                // B*T
    unsigned short* pw1 = (unsigned short*)(rstd2 + (size_t)B_ * T_);  // MH*C bf16
    unsigned short* pw2 = pw1 + (size_t)MH_ * C_;                      // C*MH bf16

    // 1) pack weights for WMMA B-operand layout
    pack_w1_kernel<<<(MH_ * C_) / 256, 256, 0, stream>>>(m1w, pw1);
    pack_w2_kernel<<<(C_ * MH_) / 256, 256, 0, stream>>>(m2w, pw2);

    // 2) LayerNorm1
    ln1_kernel<<<dim3(T_ / 256, B_), 256, 0, stream>>>(x, ln1g, ln1b, nx);

    // 3) masked average partial sums
    gctx_kernel<<<B_ * C_, 256, 0, stream>>>(nx, mask, gctx);

    // 4) gate MLP + softmax
    gate_kernel<<<1, 128, 0, stream>>>(gctx, mask, g1w, g1b, g2w, g2b, gw);

    // 5) depthwise convs + gated fusion + residual -> main_out
    conv_main_kernel<<<dim3(T_ / 256, C_, B_), 256, 0, stream>>>(
        x, mask, nx, psi_w, psi_b, wgt_w, wgt_b, ak3_w, ak3_b, ak7_w, ak7_b, gw, mainb);

    // 6) LayerNorm2 statistics
    ln2stats_kernel<<<dim3(T_ / 256, B_), 256, 0, stream>>>(mainb, mu2, rstd2);

    // 7) fused channel MLP (bf16 WMMA, async-LDS double-buffered B) -> out
    mlp_fused_kernel<<<(B_ * T_) / MROWS, 256, 0, stream>>>(
        mainb, mu2, rstd2, ln2g, ln2b, pw1, m1b, pw2, m2b, out);

    // 8) mask passthrough (second tuple output), if the harness expects it
    if (out_size >= (int)(SZ + (size_t)B_ * T_)) {
        mask_out_kernel<<<(B_ * T_ + 255) / 256, 256, 0, stream>>>(mask, out + SZ);
    }
}
